// enc_72739566125089
// MI455X (gfx1250) — matrix operand order repaired and checked by reading, not compiled, not run on hardware
//
#include <hip/hip_runtime.h>
#include <math.h>

typedef __attribute__((ext_vector_type(16))) _Float16 v16h;
typedef __attribute__((ext_vector_type(8)))  _Float16 v8h;
typedef __attribute__((ext_vector_type(4)))  _Float16 v4h;
typedef __attribute__((ext_vector_type(8)))  float    v8f;
typedef __attribute__((ext_vector_type(4)))  float    v4f;

#define DH 128   // feature width (D_IN == D_H == 128)

// ---------------- utility kernels ----------------

__global__ void k_zero_f32(float* __restrict__ p, long n) {
    long i = (long)blockIdx.x * blockDim.x + threadIdx.x;
    long stride = (long)gridDim.x * blockDim.x;
    for (; i < n; i += stride) p[i] = 0.0f;
}

__global__ void k_deg(const int* __restrict__ ei, int E, unsigned* __restrict__ cnt) {
    int e = blockIdx.x * blockDim.x + threadIdx.x;
    if (e >= E) return;
    int d = ei[E + e];
    atomicAdd(&cnt[d], 1u);
}

__global__ void k_invdeg(float* __restrict__ p, int n) {
    int i = blockIdx.x * blockDim.x + threadIdx.x;
    if (i >= n) return;
    unsigned c = ((const unsigned*)p)[i];
    p[i] = (c > 0u) ? (1.0f / (float)c) : 0.0f;
}

__global__ void k_cvt_f16(const float* __restrict__ in, _Float16* __restrict__ out, long n) {
    long i = (long)blockIdx.x * blockDim.x + threadIdx.x;
    if (i >= n) return;
    out[i] = (_Float16)in[i];
}

// Pack W[128][128] (row-major, W[k][n]) into WMMA B-fragment layout.
// Fragment (kt,nt): lane<16 -> col n=lane, j<8 holds K=kt*32+j, j>=8 holds K=kt*32+16+(j-8)
//                   lane>=16 -> col n=lane-16, j<8 holds K=kt*32+8+j, j>=8 holds K=kt*32+24+(j-8)
__global__ void k_packW(const float* __restrict__ W, _Float16* __restrict__ P) {
    int idx = blockIdx.x * blockDim.x + threadIdx.x;   // 0..16383
    if (idx >= 4096 * 4) return;
    int j    = idx & 15;
    int lane = (idx >> 4) & 31;
    int nt   = (idx >> 9) & 7;
    int kt   = idx >> 12;
    int k = kt * 32 + ((j >> 3) << 4) + ((lane >> 4) << 3) + (j & 7);
    int n = nt * 16 + (lane & 15);
    P[idx] = (_Float16)W[k * DH + n];
}

// Edge scatter: one wave (32 lanes) per edge; lane handles 4 features.
__global__ void k_scatter(const int* __restrict__ ei, int E,
                          const _Float16* __restrict__ F,
                          float* __restrict__ aggr) {
    long gid  = (long)blockIdx.x * blockDim.x + threadIdx.x;
    int  e    = (int)(gid >> 5);
    int  lane = (int)(gid & 31);
    if (e >= E) return;
    int s = ei[e];
    int d = ei[E + e];
    v4h v = *(const v4h*)(F + (long)s * DH + lane * 4);
    float* ap = aggr + (long)d * DH + lane * 4;
    atomicAdd(ap + 0, (float)v[0]);
    atomicAdd(ap + 1, (float)v[1]);
    atomicAdd(ap + 2, (float)v[2]);
    atomicAdd(ap + 3, (float)v[3]);
}

// Node update: Fout = relu( (aggr*invdeg) @ Wl + Fin @ Wr + b ), all via WMMA.
// Packed weights (2 x 32 KB) are staged into LDS once per block so the WMMA
// inner loop feeds from ds_load instead of the VMEM/L2 path.
__global__ __launch_bounds__(256)
void k_node(const float* __restrict__ aggr,
            const float* __restrict__ invdeg,
            const _Float16* __restrict__ Fin,
            const _Float16* __restrict__ Pl,
            const _Float16* __restrict__ Pr,
            const float* __restrict__ bias,
            _Float16* __restrict__ Fout,
            int N) {
    __shared__ __attribute__((aligned(32))) _Float16 sPl[DH * DH];
    __shared__ __attribute__((aligned(32))) _Float16 sPr[DH * DH];

    // cooperative staging: 2 x 32 KB, 16B chunks
    {
        uint4* sp = (uint4*)sPl;
        const uint4* gp = (const uint4*)Pl;
        for (int i = threadIdx.x; i < 2048; i += 256) sp[i] = gp[i];
        sp = (uint4*)sPr;
        gp = (const uint4*)Pr;
        for (int i = threadIdx.x; i < 2048; i += 256) sp[i] = gp[i];
    }
    __syncthreads();

    int wave = threadIdx.x >> 5;
    int lane = threadIdx.x & 31;
    long tile = (long)blockIdx.x * 8 + wave;
    int rowBase = (int)(tile * 16);
    if (rowBase >= N) return;

    int sel = lane >> 4;                 // 0: K 0-7/16-23, 1: K 8-15/24-31
    int rA  = rowBase + (lane & 15);     // row this lane feeds into A fragments
    int rAc = rA < N ? rA : N - 1;       // clamp for partial last tile
    float inv = invdeg[rAc];

    v8f acc[8];
#pragma unroll
    for (int nt = 0; nt < 8; ++nt) {
        float bv = bias[nt * 16 + (lane & 15)];
#pragma unroll
        for (int r = 0; r < 8; ++r) acc[nt][r] = bv;
    }

    const _Float16* frow = Fin  + (long)rAc * DH;
    const float*    arow = aggr + (long)rAc * DH;

#pragma unroll
    for (int kt = 0; kt < 4; ++kt) {
        // A fragment from f16 features
        v8h lo = *(const v8h*)(frow + kt * 32 + sel * 8);
        v8h hi = *(const v8h*)(frow + kt * 32 + 16 + sel * 8);
        v16h ax;
#pragma unroll
        for (int i = 0; i < 8; ++i) { ax[i] = lo[i]; ax[8 + i] = hi[i]; }

        // A fragment from f32 aggregation * (1/deg)
        v4f q0 = *(const v4f*)(arow + kt * 32 + sel * 8);
        v4f q1 = *(const v4f*)(arow + kt * 32 + sel * 8 + 4);
        v4f q2 = *(const v4f*)(arow + kt * 32 + 16 + sel * 8);
        v4f q3 = *(const v4f*)(arow + kt * 32 + 16 + sel * 8 + 4);
        v16h aa;
#pragma unroll
        for (int i = 0; i < 4; ++i) {
            aa[i]      = (_Float16)(q0[i] * inv);
            aa[4 + i]  = (_Float16)(q1[i] * inv);
            aa[8 + i]  = (_Float16)(q2[i] * inv);
            aa[12 + i] = (_Float16)(q3[i] * inv);
        }

#pragma unroll
        for (int nt = 0; nt < 8; ++nt) {
            int foff = ((kt * 8 + nt) * 32 + lane) << 4;
            v16h bl = *(const v16h*)(sPl + foff);
            v16h br = *(const v16h*)(sPr + foff);
            acc[nt] = __builtin_amdgcn_wmma_f32_16x16x32_f16(
                false, aa, false, bl, (short)0, acc[nt], false, false);
            acc[nt] = __builtin_amdgcn_wmma_f32_16x16x32_f16(
                false, ax, false, br, (short)0, acc[nt], false, false);
        }
    }

    // Epilogue: ReLU + f16 store. C layout: VGPR r -> M=r (lanes 0-15) / M=r+8.
#pragma unroll
    for (int nt = 0; nt < 8; ++nt) {
#pragma unroll
        for (int r = 0; r < 8; ++r) {
            int row = rowBase + r + sel * 8;
            if (row < N) {
                float v = acc[nt][r];
                v = v > 0.0f ? v : 0.0f;
                Fout[(long)row * DH + nt * 16 + (lane & 15)] = (_Float16)v;
            }
        }
    }
}

__global__ void k_pool(const _Float16* __restrict__ F, float* __restrict__ pool, int N) {
    int tid = blockIdx.x * blockDim.x + threadIdx.x;
    int col = tid & (DH - 1);
    int chunk = tid >> 7;
    int nchunks = (gridDim.x * blockDim.x) >> 7;
    float s = 0.0f;
    for (int r = chunk; r < N; r += nchunks) s += (float)F[(long)r * DH + col];
    atomicAdd(&pool[col], s);
}

__global__ void k_head(const float* __restrict__ pool,
                       const float* __restrict__ Wg,
                       const float* __restrict__ bg,
                       float* __restrict__ out, int N, int dlat) {
    int j = threadIdx.x;
    if (j >= dlat) return;
    float invN = 1.0f / (float)N;
    float s = bg[j];
    for (int c = 0; c < DH; ++c) s += (pool[c] * invN) * Wg[c * dlat + j];
    out[j] = 1.0f / (1.0f + expf(-s));
}

// ---------------- launch ----------------

extern "C" void kernel_launch(void* const* d_in, const int* in_sizes, int n_in,
                              void* d_out, int out_size, void* d_ws, size_t ws_size,
                              hipStream_t stream) {
    const float* x  = (const float*)d_in[0];
    const int*   ei = (const int*)d_in[1];
    const float* Wl[3] = { (const float*)d_in[2], (const float*)d_in[5], (const float*)d_in[8] };
    const float* Wr[3] = { (const float*)d_in[3], (const float*)d_in[6], (const float*)d_in[9] };
    const float* bs[3] = { (const float*)d_in[4], (const float*)d_in[7], (const float*)d_in[10] };
    const float* Wg = (const float*)d_in[11];
    const float* bg = (const float*)d_in[12];
    float* out = (float*)d_out;

    const int N = in_sizes[0] / DH;
    const int E = in_sizes[1] / 2;
    const int DLAT = out_size;

    // workspace layout (256B aligned)
    char* ws = (char*)d_ws;
    size_t off = 0;
    auto take = [&](size_t bytes) -> void* {
        void* p = ws + off;
        off = (off + bytes + 255) & ~(size_t)255;
        return p;
    };
    float*    aggr = (float*)take((size_t)N * DH * sizeof(float));
    _Float16* bufA = (_Float16*)take((size_t)N * DH * sizeof(_Float16));
    _Float16* bufB = (_Float16*)take((size_t)N * DH * sizeof(_Float16));
    float*    deg  = (float*)take((size_t)N * sizeof(float));
    _Float16* Pl   = (_Float16*)take((size_t)DH * DH * sizeof(_Float16));
    _Float16* Pr   = (_Float16*)take((size_t)DH * DH * sizeof(_Float16));
    float*    pool = (float*)take((size_t)DH * sizeof(float));
    (void)ws_size; (void)n_in;

    const int TB = 256;
    long nfeat = (long)N * DH;

    // degree (once)
    k_zero_f32<<<1024, TB, 0, stream>>>(deg, N);
    k_deg<<<(E + TB - 1) / TB, TB, 0, stream>>>(ei, E, (unsigned*)deg);
    k_invdeg<<<(N + TB - 1) / TB, TB, 0, stream>>>(deg, N);

    // x -> f16
    k_cvt_f16<<<(int)((nfeat + TB - 1) / TB), TB, 0, stream>>>(x, bufA, nfeat);

    long scatterThreads = (long)E * 32;
    int  scatterBlocks  = (int)((scatterThreads + TB - 1) / TB);
    int  tiles = (N + 15) / 16;
    int  nodeBlocks = (tiles + 7) / 8;

    for (int l = 0; l < 3; ++l) {
        const _Float16* Fin = (l & 1) ? bufB : bufA;
        _Float16*       Fout = (l & 1) ? bufA : bufB;
        k_zero_f32<<<2048, TB, 0, stream>>>(aggr, nfeat);
        k_packW<<<64, TB, 0, stream>>>(Wl[l], Pl);
        k_packW<<<64, TB, 0, stream>>>(Wr[l], Pr);
        k_scatter<<<scatterBlocks, TB, 0, stream>>>(ei, E, Fin, aggr);
        k_node<<<nodeBlocks, TB, 0, stream>>>(aggr, deg, Fin, Pl, Pr, bs[l], Fout, N);
    }

    // final features are in bufB (layers 0,1,2 -> out B,A,B)
    k_zero_f32<<<1, TB, 0, stream>>>(pool, DH);
    k_pool<<<256, TB, 0, stream>>>(bufB, pool, N);
    k_head<<<1, 64, 0, stream>>>(pool, Wg, bg, out, N, DLAT);
}
